// GCNLSTMCell_70686571758223
// MI455X (gfx1250) — matrix operand order repaired and checked by reading, not compiled, not run on hardware
//
#include <hip/hip_runtime.h>
#include <hip/hip_bf16.h>

// ---------- types ----------
typedef __attribute__((ext_vector_type(16))) __bf16        v16bf;
typedef __attribute__((ext_vector_type(8)))  float         v8f;
typedef __attribute__((ext_vector_type(4)))  float         f32x4;
typedef __attribute__((ext_vector_type(4)))  unsigned int  u32x4;
typedef __attribute__((ext_vector_type(8)))  int           i32x8;
typedef __attribute__((ext_vector_type(4)))  int           i32x4;
typedef __attribute__((ext_vector_type(4)))  unsigned short u16x4;

struct Pack32 { u32x4 lo, hi; };   // 32 bytes == v16bf

#define D_IN   512
#define HDIM   512
#define DCOMB  1024
#define FOURH  2048
#define BATCH  2

__device__ __forceinline__ unsigned short f32_to_bf16(float f) {
    unsigned int u = __float_as_uint(f);
    unsigned int r = u + 0x7FFFu + ((u >> 16) & 1u);   // round-to-nearest-even
    return (unsigned short)(r >> 16);
}
__device__ __forceinline__ float sigmoidf(float x) {
    return 1.0f / (1.0f + __expf(-x));
}

// ---------- 1) degree ----------
__global__ void k_zero_deg(int* degc, int n) {
    int i = blockIdx.x * 256 + threadIdx.x;
    if (i < n) degc[i] = 0;
}
__global__ void k_count(const int* __restrict__ ei, int* degc, int E) {
    int e = blockIdx.x * 256 + threadIdx.x;
    if (e < E) atomicAdd(&degc[ei[E + e]], 1);   // dst side
}
__global__ void k_fin_deg(const int* __restrict__ degc, float* dinv, float* selfw, int n) {
    int i = blockIdx.x * 256 + threadIdx.x;
    if (i < n) {
        float d = (float)degc[i] + 1.0f;
        dinv[i]  = __frsqrt_rn(d);
        selfw[i] = 1.0f / d;
    }
}

// ---------- 2) agg init (self-loop term) + scatter ----------
__global__ __launch_bounds__(256)
void k_init_agg(const float* __restrict__ in0, const float* __restrict__ hcur,
                const float* __restrict__ selfw, float* __restrict__ agg, int N) {
    int row  = blockIdx.x;                 // 0..BN-1
    int node = row % N;
    float w  = selfw[node];
    int d0   = threadIdx.x * 4;
    f32x4 v;
    if (d0 < D_IN) v = *(const f32x4*)(in0  + (size_t)row * D_IN + d0);
    else           v = *(const f32x4*)(hcur + (size_t)row * HDIM + (d0 - D_IN));
    *(f32x4*)(agg + (size_t)row * DCOMB + d0) = v * w;
}

__global__ __launch_bounds__(256)
void k_scatter(const float* __restrict__ in0, const float* __restrict__ hcur,
               const int* __restrict__ ei, const float* __restrict__ dinv,
               float* __restrict__ agg, int N, int E) {
    int e = blockIdx.x;
    int s = ei[e];
    int t = ei[E + e];
    float nrm = dinv[s] * dinv[t];
    int d0 = threadIdx.x * 4;
    #pragma unroll
    for (int b = 0; b < BATCH; ++b) {
        size_t rs = (size_t)b * N + s, rt = (size_t)b * N + t;
        f32x4 v;
        if (d0 < D_IN) v = *(const f32x4*)(in0  + rs * D_IN + d0);
        else           v = *(const f32x4*)(hcur + rs * HDIM + (d0 - D_IN));
        float* dst = agg + rt * DCOMB + d0;
        atomicAdd(dst + 0, v.x * nrm);
        atomicAdd(dst + 1, v.y * nrm);
        atomicAdd(dst + 2, v.z * nrm);
        atomicAdd(dst + 3, v.w * nrm);
    }
}

// ---------- 3) downconvert / transpose ----------
__global__ __launch_bounds__(256)
void k_to_bf16(const float* __restrict__ agg, unsigned short* __restrict__ abf) {
    size_t i0 = ((size_t)blockIdx.x * 256 + threadIdx.x) * 4;   // grid sized exactly
    f32x4 v = *(const f32x4*)(agg + i0);
    u16x4 r;
    r.x = f32_to_bf16(v.x); r.y = f32_to_bf16(v.y);
    r.z = f32_to_bf16(v.z); r.w = f32_to_bf16(v.w);
    *(u16x4*)(abf + i0) = r;
}

__global__ __launch_bounds__(256)
void k_transpose_W(const float* __restrict__ W, unsigned short* __restrict__ Wt) {
    int idx = blockIdx.x * 256 + threadIdx.x;       // over 2048*1024
    if (idx < FOURH * DCOMB) {
        int c  = idx >> 10;          // output column  (row of Wt)
        int kk = idx & 1023;         // K index
        Wt[idx] = f32_to_bf16(W[(size_t)kk * FOURH + c]);
    }
}

// one k-step: consume CUR (loaded one step ago), prefetch NXT for KSN,
// then shape the schedule as {2 VMEM loads, 1 WMMA} x 8, {2 DS loads}.
#define GEMM_STEP(CUR, NXT, KSN)                                              \
    {                                                                         \
        v16bf afrag = __builtin_bit_cast(v16bf, ap);                          \
        _Pragma("unroll")                                                     \
        for (int t = 0; t < 8; ++t) {                                         \
            const unsigned char* pn = WtB + bOff[t] + (KSN) * 2;              \
            NXT[t].lo = *(const u32x4*)(pn);                                  \
            NXT[t].hi = *(const u32x4*)(pn + 16);                             \
            v16bf bfrag = __builtin_bit_cast(v16bf, CUR[t]);                  \
            acc[t] = __builtin_amdgcn_wmma_f32_16x16x32_bf16(                 \
                false, afrag, false, bfrag, (short)0, acc[t], false, false);  \
        }                                                                     \
        ap.lo = *(const u32x4*)(ldsA + aoff + (KSN) * 2);                     \
        ap.hi = *(const u32x4*)(ldsA + aoff + (KSN) * 2 + 16);                \
        _Pragma("unroll")                                                     \
        for (int q = 0; q < 8; ++q) {                                         \
            __builtin_amdgcn_sched_group_barrier(0x020, 2, 0); /* VMEM rd */  \
            __builtin_amdgcn_sched_group_barrier(0x008, 1, 0); /* WMMA    */  \
        }                                                                     \
        __builtin_amdgcn_sched_group_barrier(0x100, 2, 0);     /* DS rd   */  \
    }

// ---------- 4) fused GEMM (bf16 WMMA) + LSTM epilogue ----------
__global__ __launch_bounds__(512)
void k_gemm_lstm(const unsigned short* __restrict__ abf,   // (BN,1024) bf16
                 const unsigned short* __restrict__ Wt,    // (2048,1024) bf16 (W^T)
                 const float* __restrict__ bias,           // (2048)
                 const float* __restrict__ ccur,           // (BN,512)
                 float* __restrict__ hout, float* __restrict__ cout,
                 int BN) {
    extern __shared__ unsigned char ldsA[];                // 16*1024 bf16 = 32 KB
    const int rowBase = blockIdx.x * 16;
    const int tid  = threadIdx.x;
    const int wave = tid >> 5;          // 0..15
    const int lane = tid & 31;
    const int n    = lane & 15;
    const int hi   = lane >> 4;         // lane group 0/1

    // --- stage A block (16 rows x 1024 K, bf16) into LDS with the TDM ---
    if (tid < 32) {
        unsigned long long ga =
            (unsigned long long)(uintptr_t)(abf + (size_t)rowBase * DCOMB);
        u32x4 g0;
        g0.x = 1u;                                        // count = 1 valid descriptor
        g0.y = 0u;                                        // lds_addr = base of dyn LDS
        g0.z = (unsigned int)(ga & 0xFFFFFFFFull);        // global_addr[31:0]
        g0.w = (unsigned int)((ga >> 32) & 0x01FFFFFFull) // global_addr[56:32]
               | 0x80000000u;                             // type = 2 ("image")
        i32x8 g1;
        g1[0] = (int)(1u << 16);      // data_size = 1 (2 bytes/elem)
        g1[1] = (int)(1024u << 16);   // tensor_dim0 = 1024 (bits 63:48)
        g1[2] = (int)(16u << 16);     // tensor_dim1 = 16   (bits 95:80)
        g1[3] = (int)(1024u << 16);   // tile_dim0 = 1024   (bits 127:112)
        g1[4] = 16;                   // tile_dim1 = 16     (bits 143:128)
        g1[5] = 1024;                 // tensor_dim0_stride = 1024 elems
        g1[6] = 0;
        g1[7] = 0;
        i32x4 gz = {0, 0, 0, 0};
#if __clang_major__ >= 23
        i32x8 gz8 = {0, 0, 0, 0, 0, 0, 0, 0};
        __builtin_amdgcn_tensor_load_to_lds(g0, g1, gz, gz, gz8, 0);
#else
        __builtin_amdgcn_tensor_load_to_lds(g0, g1, gz, gz, 0);
#endif
        __builtin_amdgcn_s_wait_tensorcnt(0);
    }
    __syncthreads();

    // --- per-wave column tiles: {i,f,o,g} x {2 tiles of 16} at matching h ---
    const int hbase = wave * 32;                 // h-range [hbase, hbase+32)
    int bOff[8];                                 // byte offsets into Wt per tile
    #pragma unroll
    for (int t = 0; t < 8; ++t) {
        int col = (t >> 1) * HDIM + hbase + (t & 1) * 16 + n;   // output column
        bOff[t] = (col * DCOMB + hi * 16) * 2;
    }
    const int aoff = (n * DCOMB + hi * 16) * 2;  // LDS byte offset (row = n)

    v8f acc[8] = {};
    const unsigned char* WtB = (const unsigned char*)Wt;

    // --- prologue: load k-step 0 fragments ---
    Pack32 bufA[8], bufB[8];
    #pragma unroll
    for (int t = 0; t < 8; ++t) {
        bufA[t].lo = *(const u32x4*)(WtB + bOff[t]);
        bufA[t].hi = *(const u32x4*)(WtB + bOff[t] + 16);
    }
    Pack32 ap;
    ap.lo = *(const u32x4*)(ldsA + aoff);
    ap.hi = *(const u32x4*)(ldsA + aoff + 16);

    // --- ping-pong pipelined K loop: two k-steps per iteration ---
    #pragma unroll 1
    for (int ks = 0; ks < DCOMB; ks += 64) {
        const int ks1 = ks + 32;                                       // < DCOMB
        const int ks2 = (ks + 64 < DCOMB) ? (ks + 64) : (DCOMB - 32);  // clamped
        GEMM_STEP(bufA, bufB, ks1);
        GEMM_STEP(bufB, bufA, ks2);
    }

    // --- LSTM epilogue (gates are wave-local by construction) ---
    #pragma unroll
    for (int tt = 0; tt < 2; ++tt) {
        int c  = hbase + tt * 16 + n;            // h index 0..511
        float bi = bias[c];
        float bfv = bias[c + HDIM];
        float bo = bias[c + 2 * HDIM];
        float bg = bias[c + 3 * HDIM];
        #pragma unroll
        for (int r = 0; r < 8; ++r) {
            int m   = r + hi * 8;
            int row = rowBase + m;
            if (row < BN) {
                float vi = sigmoidf(acc[0 * 2 + tt][r] + bi);
                float vf = sigmoidf(acc[1 * 2 + tt][r] + bfv);
                float vo = sigmoidf(acc[2 * 2 + tt][r] + bo);
                float vg = tanhf   (acc[3 * 2 + tt][r] + bg);
                float cc = ccur[(size_t)row * HDIM + c];
                float cn = vf * cc + vi * vg;
                float hn = vo * tanhf(cn);
                hout[(size_t)row * HDIM + c] = hn;
                cout[(size_t)row * HDIM + c] = cn;
            }
        }
    }
}

// ---------- host ----------
extern "C" void kernel_launch(void* const* d_in, const int* in_sizes, int n_in,
                              void* d_out, int out_size, void* d_ws, size_t ws_size,
                              hipStream_t stream) {
    const float* in0  = (const float*)d_in[0];   // (B,N,512)
    const float* hcur = (const float*)d_in[1];   // (B,N,512)
    const float* ccur = (const float*)d_in[2];   // (B,N,512)
    const int*   ei   = (const int*)  d_in[3];   // (2,E)
    const float* W    = (const float*)d_in[4];   // (1024,2048)
    const float* bias = (const float*)d_in[5];   // (2048)

    const int N  = in_sizes[1] / (BATCH * HDIM);
    const int E  = in_sizes[3] / 2;
    const int BN = BATCH * N;

    // workspace carve-up (256B aligned)
    unsigned char* ws = (unsigned char*)d_ws;
    size_t off = 0;
    auto carve = [&](size_t bytes) {
        size_t o = off;
        off = (off + bytes + 255) & ~(size_t)255;
        return o;
    };
    int*            degc  = (int*)           (ws + carve((size_t)N * 4));
    float*          dinv  = (float*)         (ws + carve((size_t)N * 4));
    float*          selfw = (float*)         (ws + carve((size_t)N * 4));
    float*          agg   = (float*)         (ws + carve((size_t)BN * DCOMB * 4));
    unsigned short* abf   = (unsigned short*)(ws + carve((size_t)BN * DCOMB * 2));
    unsigned short* Wt    = (unsigned short*)(ws + carve((size_t)FOURH * DCOMB * 2));
    (void)ws_size; (void)n_in;

    float* hout = (float*)d_out;
    float* cout = hout + (size_t)BN * HDIM;
    (void)out_size;

    // 1) degrees
    k_zero_deg<<<(N + 255) / 256, 256, 0, stream>>>(degc, N);
    k_count   <<<(E + 255) / 256, 256, 0, stream>>>(ei, degc, E);
    k_fin_deg <<<(N + 255) / 256, 256, 0, stream>>>(degc, dinv, selfw, N);

    // 2) aggregate
    k_init_agg<<<BN, 256, 0, stream>>>(in0, hcur, selfw, agg, N);
    k_scatter <<<E, 256, 0, stream>>>(in0, hcur, ei, dinv, agg, N, E);

    // 3) precision conversion
    k_to_bf16     <<<(int)(((size_t)BN * DCOMB) / 1024), 256, 0, stream>>>(agg, abf);
    k_transpose_W <<<(FOURH * DCOMB + 255) / 256, 256, 0, stream>>>(W, Wt);

    // 4) fused WMMA GEMM + LSTM
    k_gemm_lstm<<<BN / 16, 512, 16 * DCOMB * 2, stream>>>(
        abf, Wt, bias, ccur, hout, cout, BN);
}
